// HeteroGraphMultiModalModel_22557168239254
// MI455X (gfx1250) — matrix operand order repaired and checked by reading, not compiled, not run on hardware
//
#include <hip/hip_runtime.h>
#include <math.h>

#define HID   256
#define HEADS 4
#define KP    25          // label prototypes
#define LN_EPS 1e-5f
#define NEG_SLOPE 0.2f

typedef __attribute__((ext_vector_type(16))) __bf16 v16bf;
typedef __attribute__((ext_vector_type(8)))  float  v8f;

// ---------- helpers ----------
__device__ __forceinline__ unsigned f2ord(float f) {
    unsigned u = __float_as_uint(f);
    return (u & 0x80000000u) ? ~u : (u | 0x80000000u);
}
__device__ __forceinline__ float ord2f(unsigned u) {
    return (u & 0x80000000u) ? __uint_as_float(u & 0x7fffffffu)
                             : __uint_as_float(~u);
}
#define NEGINF_ORD 0x007FFFFFu   // f2ord(-inf)

__device__ __forceinline__ float block_reduce_sum256(float v, float* red) {
    int t = threadIdx.x;
    red[t] = v; __syncthreads();
    for (int s = 128; s > 0; s >>= 1) {
        if (t < s) red[t] += red[t + s];
        __syncthreads();
    }
    float r = red[0]; __syncthreads();
    return r;
}

// ---------- conversions / inits ----------
__global__ void f32_to_bf16_kernel(const float* __restrict__ in, __bf16* __restrict__ out, int n) {
    int i = blockIdx.x * blockDim.x + threadIdx.x;
    if (i < n) out[i] = (__bf16)in[i];
}
__global__ void init_softmax_state_kernel(unsigned* mx, float* sm, int n) {
    int i = blockIdx.x * blockDim.x + threadIdx.x;
    if (i < n) { mx[i] = NEGINF_ORD; sm[i] = 0.0f; }
}
__global__ void zero_f32_kernel(float* p, int n) {
    int i = blockIdx.x * blockDim.x + threadIdx.x;
    if (i < n) p[i] = 0.0f;
}
__global__ void zero_i32_kernel(int* p, int n) {
    int i = blockIdx.x * blockDim.x + threadIdx.x;
    if (i < n) p[i] = 0;
}
__global__ void init_bias_rows_kernel(const float* __restrict__ bias, float* __restrict__ out) {
    out[blockIdx.x * HID + threadIdx.x] = bias[threadIdx.x];
}

// ---------- bf16 WMMA GEMM:  Y[M,256] = A[M,256] @ W[256,256] (+bias) ----------
// One 16x16 output tile per wave, K-steps of 32 via v_wmma_f32_16x16x32_bf16.
__global__ void gemm_bf16_256_kernel(const __bf16* __restrict__ A,
                                     const __bf16* __restrict__ W,   // row-major [k][n]
                                     const float*  __restrict__ bias, // nullable
                                     float* __restrict__ Y, int M) {
    const int lane = threadIdx.x & 31;
    const int wave = threadIdx.x >> 5;              // 8 waves / block
    const int tile = blockIdx.x * 8 + wave;
    const int tilesN = HID / 16;                    // 16
    const int totalTiles = (M >> 4) * tilesN;
    if (tile >= totalTiles) return;                 // wave-uniform guard
    const int tm = tile / tilesN, tn = tile % tilesN;
    const int hs  = lane >> 4;                      // half-select
    const int row = tm * 16 + (lane & 15);          // A row for this lane

    v8f acc = {};
    for (int k0 = 0; k0 < HID; k0 += 32) {
        v16bf a, b;
        // A 16x32 bf16 layout: elem e -> vg=e/2, half=e%2,
        // K = 2*vg + (vg>=4 ? 8 : 0) + half + 8*hs   (ISA 7.12.2)
        #pragma unroll
        for (int e = 0; e < 16; ++e) {
            int vg = e >> 1, half = e & 1;
            int kk = 2 * vg + ((vg >= 4) ? 8 : 0) + half + 8 * hs;
            a[e] = A[row * HID + k0 + kk];
        }
        // B 32x16 bf16: lane holds column N = tn*16 + lane%16; elems are K in the same order as A
        #pragma unroll
        for (int e = 0; e < 16; ++e) {
            int kb = k0 + e + ((e >= 8) ? 8 : 0) + 8 * hs;
            b[e] = W[kb * HID + tn * 16 + (lane & 15)];
        }

        acc = __builtin_amdgcn_wmma_f32_16x16x32_bf16(
                  false, a, false, b, (short)0, acc, false, false);
    }
    // C/D layout: lane -> N = lane%16; VGPR v -> M = v + 8*hs
    const int n = tn * 16 + (lane & 15);
    const float bn = bias ? bias[n] : 0.0f;
    #pragma unroll
    for (int v = 0; v < 8; ++v) {
        int m = tm * 16 + v + 8 * hs;
        Y[m * HID + n] = acc[v] + bn;
    }
}

// ---------- GAT node attention coefficients (wave-reduce per head) ----------
__global__ void att_node_kernel(const float* __restrict__ xh,
                                const float* __restrict__ att_s,
                                const float* __restrict__ att_d,
                                float* __restrict__ a_src, float* __restrict__ a_dst) {
    const int n = blockIdx.x;                       // block = 128 threads = 4 waves
    const int h = threadIdx.x >> 5, l = threadIdx.x & 31;
    float x0 = xh[n * HID + h * 64 + l];
    float x1 = xh[n * HID + h * 64 + 32 + l];
    float ss = x0 * att_s[h * 64 + l] + x1 * att_s[h * 64 + 32 + l];
    float sd = x0 * att_d[h * 64 + l] + x1 * att_d[h * 64 + 32 + l];
    #pragma unroll
    for (int off = 16; off > 0; off >>= 1) {
        ss += __shfl_down(ss, off);
        sd += __shfl_down(sd, off);
    }
    if (l == 0) { a_src[n * HEADS + h] = ss; a_dst[n * HEADS + h] = sd; }
}

// ---------- GAT edge softmax (3 passes) ----------
__global__ void edge_score_kernel(const int* __restrict__ ei,
                                  const float* __restrict__ a_src,
                                  const float* __restrict__ a_dst,
                                  float* __restrict__ e_s, unsigned* __restrict__ segmax, int E) {
    int idx = blockIdx.x * blockDim.x + threadIdx.x;
    if (idx >= E * HEADS) return;
    int e = idx >> 2, h = idx & 3;
    int s = ei[e], d = ei[E + e];
    float v = a_src[s * HEADS + h] + a_dst[d * HEADS + h];
    v = (v > 0.0f) ? v : NEG_SLOPE * v;             // leaky_relu
    e_s[idx] = v;
    atomicMax(&segmax[d * HEADS + h], f2ord(v));
}
__global__ void edge_exp_kernel(const int* __restrict__ ei,
                                float* __restrict__ e_s,          // in: score, out: exp (in-place)
                                const unsigned* __restrict__ segmax,
                                float* __restrict__ segsum, int E) {
    int idx = blockIdx.x * blockDim.x + threadIdx.x;
    if (idx >= E * HEADS) return;
    int e = idx >> 2, h = idx & 3;
    int d = ei[E + e];
    float ex = expf(e_s[idx] - ord2f(segmax[d * HEADS + h]));
    e_s[idx] = ex;
    atomicAdd(&segsum[d * HEADS + h], ex);
}
__global__ void scatter_ehr_kernel(const int* __restrict__ ei,
                                   const float* __restrict__ xh,
                                   const float* __restrict__ e_exp,
                                   const float* __restrict__ segsum,
                                   float* __restrict__ out2, int E) {
    int e = blockIdx.x, c = threadIdx.x, h = c >> 6;
    int s = ei[e], d = ei[E + e];
    float alpha = e_exp[e * HEADS + h] / (segsum[d * HEADS + h] + 1e-16f);
    atomicAdd(&out2[d * HID + c], xh[s * HID + c] * alpha);
}

// ---------- time-weighted cxr->ehr ----------
__global__ void cxr_time_max_kernel(const int* __restrict__ eic, const float* __restrict__ ct,
                                    unsigned* __restrict__ wmax, int Ec) {
    int e = blockIdx.x * blockDim.x + threadIdx.x;
    if (e >= Ec) return;
    int d = eic[Ec + e];
    float s = ct[e] * (1.0f / (0.5f + 1e-8f));
    atomicMax(&wmax[d], f2ord(s));
}
__global__ void cxr_time_sum_kernel(const int* __restrict__ eic, const float* __restrict__ ct,
                                    const unsigned* __restrict__ wmax,
                                    float* __restrict__ wexp, float* __restrict__ wsum,
                                    int* __restrict__ has, int Ec) {
    int e = blockIdx.x * blockDim.x + threadIdx.x;
    if (e >= Ec) return;
    int d = eic[Ec + e];
    float s = ct[e] * (1.0f / (0.5f + 1e-8f));
    float ex = expf(s - ord2f(wmax[d]));
    wexp[e] = ex;
    atomicAdd(&wsum[d], ex);
    has[d] = 1;
}
__global__ void scatter_cxr_kernel(const int* __restrict__ eic, const float* __restrict__ ycxr,
                                   const float* __restrict__ wexp, const float* __restrict__ wsum,
                                   float* __restrict__ out3, int Ec) {
    int e = blockIdx.x, c = threadIdx.x;
    int s = eic[e], d = eic[Ec + e];
    float w = wexp[e] / (wsum[d] + 1e-16f);
    atomicAdd(&out3[d * HID + c], ycxr[s * HID + c] * w);
}

// ---------- token stack + LayerNorm -> bf16 tokens ----------
__global__ void tokens_ln_kernel(const float* __restrict__ x_ehr,
                                 const float* __restrict__ mee,
                                 const float* __restrict__ mce,
                                 const float* __restrict__ g, const float* __restrict__ b,
                                 __bf16* __restrict__ tok) {
    __shared__ float red[256];
    int n = blockIdx.x / 3, t = blockIdx.x % 3, c = threadIdx.x;
    const float* src = (t == 0) ? x_ehr : (t == 1) ? mee : mce;
    float v = src[n * HID + c];
    float mu  = block_reduce_sum256(v, red) * (1.0f / HID);
    float d   = v - mu;
    float var = block_reduce_sum256(d * d, red) * (1.0f / HID);
    float y = d * rsqrtf(var + LN_EPS) * g[c] + b[c];
    tok[blockIdx.x * HID + c] = (__bf16)y;
}

// ---------- row l2norm (in place) ----------
__global__ void l2norm_rows_kernel(float* __restrict__ X) {
    __shared__ float red[256];
    int r = blockIdx.x, c = threadIdx.x;
    float v = X[r * HID + c];
    float ss = block_reduce_sum256(v * v, red);
    X[r * HID + c] = v / fmaxf(sqrtf(ss), 1e-12f);
}

// ---------- Q = l2norm(label_proto @ pq_w + pq_b), tiny GEMM ----------
__global__ void qlbl_kernel(const float* __restrict__ lp, const float* __restrict__ w,
                            const float* __restrict__ bias, float* __restrict__ Qn) {
    __shared__ float red[256];
    int k = blockIdx.x, c = threadIdx.x;
    float s = bias[c];
    for (int j = 0; j < HID; ++j) s += lp[k * HID + j] * w[j * HID + c];
    float ss = block_reduce_sum256(s * s, red);
    Qn[k * HID + c] = s / fmaxf(sqrtf(ss), 1e-12f);
}

// ---------- per-node cross attention + LN2 ----------
__global__ void attn_z_kernel(const float* __restrict__ Kt, const float* __restrict__ Vt,
                              const float* __restrict__ Qn, const int* __restrict__ has_cxr,
                              const float* __restrict__ g2, const float* __restrict__ b2,
                              float* __restrict__ out_z) {
    __shared__ float sK[3][HID];
    __shared__ float sV[3][HID];
    __shared__ float sQ[KP * HID];
    __shared__ float sZ[KP * HID];
    __shared__ float sAttn[3][KP];
    __shared__ float red[256];
    int n = blockIdx.x, c = threadIdx.x;
    #pragma unroll
    for (int m = 0; m < 3; ++m) {
        sK[m][c] = Kt[(n * 3 + m) * HID + c];
        sV[m][c] = Vt[(n * 3 + m) * HID + c];
    }
    for (int k = 0; k < KP; ++k) sQ[k * HID + c] = Qn[k * HID + c];
    __syncthreads();

    if (c < KP) {
        float s[3];
        #pragma unroll
        for (int m = 0; m < 3; ++m) {
            float acc = 0.0f;
            for (int j = 0; j < HID; ++j) acc += sK[m][j] * sQ[c * HID + j];
            s[m] = acc;
        }
        bool has = (has_cxr[n] != 0);
        float mx = fmaxf(s[0], s[1]);
        if (has) mx = fmaxf(mx, s[2]);
        float e0 = expf(s[0] - mx), e1 = expf(s[1] - mx);
        float e2 = has ? expf(s[2] - mx) : 0.0f;
        float inv = 1.0f / (e0 + e1 + e2);
        sAttn[0][c] = e0 * inv; sAttn[1][c] = e1 * inv; sAttn[2][c] = e2 * inv;
    }
    __syncthreads();

    for (int k = 0; k < KP; ++k)
        sZ[k * HID + c] = sAttn[0][k] * sV[0][c] + sAttn[1][k] * sV[1][c]
                        + sAttn[2][k] * sV[2][c];
    __syncthreads();

    for (int k = 0; k < KP; ++k) {
        float v = sZ[k * HID + c];
        float mu  = block_reduce_sum256(v, red) * (1.0f / HID);
        float d   = v - mu;
        float var = block_reduce_sum256(d * d, red) * (1.0f / HID);
        out_z[(n * KP + k) * HID + c] = d * rsqrtf(var + LN_EPS) * g2[c] + b2[c];
    }
}

// ---------- launch ----------
extern "C" void kernel_launch(void* const* d_in, const int* in_sizes, int n_in,
                              void* d_out, int out_size, void* d_ws, size_t ws_size,
                              hipStream_t stream) {
    const float* x_ehr    = (const float*)d_in[0];
    const float* x_cxr    = (const float*)d_in[1];
    const float* cxr_time = (const float*)d_in[2];
    const float* lbl_pro  = (const float*)d_in[3];
    const float* gat_w    = (const float*)d_in[4];
    const float* att_src  = (const float*)d_in[5];
    const float* att_dst  = (const float*)d_in[6];
    const float* gat_bias = (const float*)d_in[7];
    const float* time_w   = (const float*)d_in[8];
    const float* pk_w     = (const float*)d_in[9];
    const float* pk_b     = (const float*)d_in[10];
    const float* pq_w     = (const float*)d_in[11];
    const float* pq_b     = (const float*)d_in[12];
    const float* pv_w     = (const float*)d_in[13];
    const float* pv_b     = (const float*)d_in[14];
    const float* ln1_g    = (const float*)d_in[15];
    const float* ln1_b    = (const float*)d_in[16];
    const float* ln2_g    = (const float*)d_in[17];
    const float* ln2_b    = (const float*)d_in[18];
    const int*   ei_ehr   = (const int*)d_in[19];
    const int*   ei_cxr   = (const int*)d_in[20];

    const int N    = in_sizes[0] / HID;
    const int Ncx  = in_sizes[1] / HID;
    const int E    = in_sizes[19] / 2;
    const int Ec   = in_sizes[20] / 2;

    float* out_z = (float*)d_out;                 // [N, KP, HID]
    float* out2  = out_z + (size_t)N * KP * HID;  // msg_ehr_ehr [N, HID]
    float* out3  = out2  + (size_t)N * HID;       // msg_cxr_ehr [N, HID]

    // workspace carve-up
    size_t off = 0;
    auto carve = [&](size_t bytes) -> char* {
        char* p = (char*)d_ws + off;
        off += (bytes + 255) & ~(size_t)255;
        return p;
    };
    __bf16*  xe_bf   = (__bf16*) carve((size_t)N   * HID * 2);
    __bf16*  xc_bf   = (__bf16*) carve((size_t)Ncx * HID * 2);
    __bf16*  gw_bf   = (__bf16*) carve((size_t)HID * HID * 2);
    __bf16*  tw_bf   = (__bf16*) carve((size_t)HID * HID * 2);
    __bf16*  pkw_bf  = (__bf16*) carve((size_t)HID * HID * 2);
    __bf16*  pvw_bf  = (__bf16*) carve((size_t)HID * HID * 2);
    float*   xh      = (float*)  carve((size_t)N   * HID * 4);
    float*   ycxr    = (float*)  carve((size_t)Ncx * HID * 4);
    float*   a_src   = (float*)  carve((size_t)N * HEADS * 4);
    float*   a_dst   = (float*)  carve((size_t)N * HEADS * 4);
    unsigned* segmax = (unsigned*)carve((size_t)N * HEADS * 4);
    float*   segsum  = (float*)  carve((size_t)N * HEADS * 4);
    float*   e_s     = (float*)  carve((size_t)E * HEADS * 4);   // score -> exp, in-place
    unsigned* wmax   = (unsigned*)carve((size_t)N * 4);
    float*   wsum    = (float*)  carve((size_t)N * 4);
    float*   wexp    = (float*)  carve((size_t)Ec * 4);
    int*     has     = (int*)    carve((size_t)N * 4);
    __bf16*  tok_bf  = (__bf16*) carve((size_t)N * 3 * HID * 2);
    float*   Kt      = (float*)  carve((size_t)N * 3 * HID * 4);
    float*   Vt      = (float*)  carve((size_t)N * 3 * HID * 4);
    float*   Qn      = (float*)  carve((size_t)KP * HID * 4);
    (void)ws_size; (void)n_in; (void)out_size;

    const int TPB = 256;
    auto cdiv = [](int a, int b) { return (a + b - 1) / b; };

    // 1) bf16 conversions
    f32_to_bf16_kernel<<<cdiv(N * HID, TPB), TPB, 0, stream>>>(x_ehr, xe_bf, N * HID);
    f32_to_bf16_kernel<<<cdiv(Ncx * HID, TPB), TPB, 0, stream>>>(x_cxr, xc_bf, Ncx * HID);
    f32_to_bf16_kernel<<<cdiv(HID * HID, TPB), TPB, 0, stream>>>(gat_w, gw_bf, HID * HID);
    f32_to_bf16_kernel<<<cdiv(HID * HID, TPB), TPB, 0, stream>>>(time_w, tw_bf, HID * HID);
    f32_to_bf16_kernel<<<cdiv(HID * HID, TPB), TPB, 0, stream>>>(pk_w, pkw_bf, HID * HID);
    f32_to_bf16_kernel<<<cdiv(HID * HID, TPB), TPB, 0, stream>>>(pv_w, pvw_bf, HID * HID);

    // 2) WMMA GEMMs: xh = x_ehr @ gat_w ; ycxr = x_cxr @ time_w
    {
        int tiles = (N >> 4) * (HID / 16);
        gemm_bf16_256_kernel<<<cdiv(tiles, 8), 256, 0, stream>>>(xe_bf, gw_bf, nullptr, xh, N);
        int tilesc = (Ncx >> 4) * (HID / 16);
        gemm_bf16_256_kernel<<<cdiv(tilesc, 8), 256, 0, stream>>>(xc_bf, tw_bf, nullptr, ycxr, Ncx);
    }

    // 3) attention coefficients
    att_node_kernel<<<N, 128, 0, stream>>>(xh, att_src, att_dst, a_src, a_dst);

    // 4) state init
    init_softmax_state_kernel<<<cdiv(N * HEADS, TPB), TPB, 0, stream>>>(segmax, segsum, N * HEADS);
    init_softmax_state_kernel<<<cdiv(N, TPB), TPB, 0, stream>>>(wmax, wsum, N);
    zero_i32_kernel<<<cdiv(N, TPB), TPB, 0, stream>>>(has, N);
    init_bias_rows_kernel<<<N, HID, 0, stream>>>(gat_bias, out2);   // out2 starts at +gat_bias
    zero_f32_kernel<<<cdiv(N * HID, TPB), TPB, 0, stream>>>(out3, N * HID);

    // 5) GAT edge softmax + scatter
    edge_score_kernel<<<cdiv(E * HEADS, TPB), TPB, 0, stream>>>(ei_ehr, a_src, a_dst, e_s, segmax, E);
    edge_exp_kernel<<<cdiv(E * HEADS, TPB), TPB, 0, stream>>>(ei_ehr, e_s, segmax, segsum, E);
    scatter_ehr_kernel<<<E, HID, 0, stream>>>(ei_ehr, xh, e_s, segsum, out2, E);

    // 6) cxr time softmax + scatter
    cxr_time_max_kernel<<<cdiv(Ec, TPB), TPB, 0, stream>>>(ei_cxr, cxr_time, wmax, Ec);
    cxr_time_sum_kernel<<<cdiv(Ec, TPB), TPB, 0, stream>>>(ei_cxr, cxr_time, wmax, wexp, wsum, has, Ec);
    scatter_cxr_kernel<<<Ec, HID, 0, stream>>>(ei_cxr, ycxr, wexp, wsum, out3, Ec);

    // 7) token LayerNorm -> bf16
    tokens_ln_kernel<<<N * 3, HID, 0, stream>>>(x_ehr, out2, out3, ln1_g, ln1_b, tok_bf);

    // 8) K/V projections (WMMA), 9) l2norm K
    {
        int tiles = ((N * 3) >> 4) * (HID / 16);
        gemm_bf16_256_kernel<<<cdiv(tiles, 8), 256, 0, stream>>>(tok_bf, pkw_bf, pk_b, Kt, N * 3);
        gemm_bf16_256_kernel<<<cdiv(tiles, 8), 256, 0, stream>>>(tok_bf, pvw_bf, pv_b, Vt, N * 3);
    }
    l2norm_rows_kernel<<<N * 3, HID, 0, stream>>>(Kt);

    // 10) label queries
    qlbl_kernel<<<KP, HID, 0, stream>>>(lbl_pro, pq_w, pq_b, Qn);

    // 11) masked cross attention + LN2 -> z
    attn_z_kernel<<<N, HID, 0, stream>>>(Kt, Vt, Qn, has, ln2_g, ln2_b, out_z);
}
